// Transformer_26757646254304
// MI455X (gfx1250) — matrix operand order repaired and checked by reading, not compiled, not run on hardware
//
#include <hip/hip_runtime.h>
#include <math.h>

#define DIM    512
#define DEPTH  4
#define BLOCK  64
#define HEADS  8
#define HDIM   64
#define VOCAB  32000
#define BATCH  2
#define SEQ    2048
#define MROWS  (BATCH*SEQ)   // 4096
#define EPS    1e-6f

typedef __attribute__((ext_vector_type(8)))  __bf16 v8bf;
typedef __attribute__((ext_vector_type(16))) __bf16 v16bf;
typedef __attribute__((ext_vector_type(8)))  float  v8f;
typedef int v4i_gcc __attribute__((vector_size(16)));   // matches builtin param

#if __has_builtin(__builtin_amdgcn_global_load_async_to_lds_b128)
#define ASYNC_LDS 1
#else
#define ASYNC_LDS 0
#endif

// ---------------------------------------------------------------- helpers ---

static __device__ inline unsigned short f2bfu(float f) {
    union { float f; unsigned u; } a; a.f = f;
    unsigned u = a.u;
    u += 0x7FFFu + ((u >> 16) & 1u);           // round-to-nearest-even
    return (unsigned short)(u >> 16);
}

static __device__ inline __bf16 f2bf(float f) {
    unsigned short s = f2bfu(f);
    union { unsigned short s; __bf16 b; } c; c.s = s;
    return c.b;
}

static __device__ inline float gelu_tanh(float x) {
    float x3 = x * x * x;
    return 0.5f * x * (1.0f + tanhf(0.7978845608028654f * (x + 0.044715f * x3)));
}

// async 16-byte global->LDS copy (ASYNCcnt tracked); fallback via VGPRs.
static __device__ inline void cp16(__bf16* lds, const __bf16* g) {
#if ASYNC_LDS
    __builtin_amdgcn_global_load_async_to_lds_b128(
        (__attribute__((address_space(1))) v4i_gcc*)(__bf16*)g,
        (__attribute__((address_space(3))) v4i_gcc*)lds, 0, 0);
#else
    *(uint4*)lds = *(const uint4*)g;
#endif
}

template<int N> static __device__ inline void wait_asynccnt() {
#if __has_builtin(__builtin_amdgcn_s_wait_asynccnt)
    __builtin_amdgcn_s_wait_asynccnt((unsigned short)N);
#else
    asm volatile("s_wait_asynccnt %0" :: "i"(N) : "memory");
#endif
}

// A fragment: 16x32 bf16, row m = lane&15, K = kb..kb+15 contiguous,
// kb = (lane>=16)*16.  tile row-major, leading dim ld (elements).
static __device__ inline v16bf load_a_frag(const __bf16* tile, int mBase, int ld,
                                           int kOff, int lane) {
    int m  = mBase + (lane & 15);
    int kb = (lane >> 4) << 4;
    return *(const v16bf*)(tile + m * ld + kOff + kb);
}

// B fragment: 32x16 bf16 stored row-major [n][k]: n = lane&15,
// K = kOff + (lane>=16)*16 .. +15 contiguous.
static __device__ inline v16bf load_b_frag(const __bf16* tile, int nBase, int ld,
                                           int kOff, int lane) {
    int n  = nBase + (lane & 15);
    int ks = kOff + ((lane >> 4) << 4);
    return *(const v16bf*)(tile + n * ld + ks);
}

static __device__ inline v8f wmma_bf16(v16bf a, v16bf b, v8f c) {
    return __builtin_amdgcn_wmma_f32_16x16x32_bf16(false, a, false, b,
                                                   (short)0, c, false, false);
}

// ------------------------------------------------------- fp32 -> bf16 pack ---

__global__ __launch_bounds__(256) void convert_bf16_kernel(
    const float* __restrict__ src, unsigned int* __restrict__ dst, int n4)
{
    int stride = gridDim.x * 256;
    for (int i = blockIdx.x * 256 + threadIdx.x; i < n4; i += stride) {
        float4 f = *(const float4*)(src + (size_t)i * 4);
        uint2 p;
        p.x = (unsigned)f2bfu(f.x) | ((unsigned)f2bfu(f.y) << 16);
        p.y = (unsigned)f2bfu(f.z) | ((unsigned)f2bfu(f.w) << 16);
        *(uint2*)(dst + (size_t)i * 2) = p;
    }
}

// ------------------------------------------------------------ embed + pos ---

__global__ __launch_bounds__(256) void embed_pe_kernel(
    const int* __restrict__ seq, const float* __restrict__ emb,
    float* __restrict__ x)
{
    int row = blockIdx.x;                 // 0..MROWS-1
    int s   = row & (SEQ - 1);
    int tok = seq[row];
    for (int d = threadIdx.x; d < DIM; d += 256) {
        int i = d >> 1;
        float dv  = __expf((float)(2 * i) * (-9.210340371976184f / (float)DIM));
        float ang = (float)s * dv;
        float pe  = (d & 1) ? __cosf(ang) : __sinf(ang);
        x[(size_t)row * DIM + d] = emb[(size_t)tok * DIM + d] + pe;
    }
}

// -------------------------------------------------------- rmsnorm -> bf16 ---

__global__ __launch_bounds__(256) void rmsnorm_kernel(
    const float* __restrict__ x, const float* __restrict__ w,
    const float* __restrict__ b, __bf16* __restrict__ out)
{
    int row = blockIdx.x;
    int tid = threadIdx.x, lane = tid & 31;
    const float* xr = x + (size_t)row * DIM;
    float v0 = xr[tid], v1 = xr[tid + 256];
    float s = v0 * v0 + v1 * v1;
    #pragma unroll
    for (int off = 16; off >= 1; off >>= 1) s += __shfl_xor(s, off, 32);
    __shared__ float red[8];
    if (lane == 0) red[tid >> 5] = s;
    __syncthreads();
    float tot = 0.f;
    #pragma unroll
    for (int i = 0; i < 8; ++i) tot += red[i];
    float inv = rsqrtf(tot / (float)DIM + EPS);
    out[(size_t)row * DIM + tid]       = f2bf(v0 * inv * w[tid]       + b[tid]);
    out[(size_t)row * DIM + tid + 256] = f2bf(v1 * inv * w[tid + 256] + b[tid + 256]);
}

// -------------------------------------------------------------- WMMA GEMM ---
// C[M,N] = epi( A[M,K]bf16 @ W[N,K]bf16^T ), 128x64 block tile, K-step 64,
// double-buffered async global->LDS tile loads, 4 waves x (32m x 64n):
// each B fragment is reused by two WMMAs (two A fragments per wave).

template<bool BIAS, bool GELU, bool RES, bool OUTBF>
__global__ __launch_bounds__(128) void gemm_bf16_kernel(
    const __bf16* __restrict__ A, const __bf16* __restrict__ W,
    const float* __restrict__ bias, const float* __restrict__ residual,
    float* __restrict__ C, __bf16* __restrict__ Cb, int N, int K)
{
    __shared__ alignas(64) __bf16 At[2][128 * 64];   // 2 x 16 KiB
    __shared__ alignas(64) __bf16 Bt[2][64 * 64];    // 2 x  8 KiB
    int tid = threadIdx.x, lane = tid & 31, w = tid >> 5;
    int nBase = blockIdx.x * 64, mBase = blockIdx.y * 128;

    auto issueTiles = [&](int k0, int buf) {
        #pragma unroll
        for (int v = 0; v < 8; ++v) {                 // A: 1024 x 16B
            int e = tid + v * 128;
            int r = e >> 3, c8 = (e & 7) << 3;
            cp16(&At[buf][r * 64 + c8], A + (size_t)(mBase + r) * K + k0 + c8);
        }
        #pragma unroll
        for (int v = 0; v < 4; ++v) {                 // B: 512 x 16B
            int e = tid + v * 128;
            int r = e >> 3, c8 = (e & 7) << 3;
            cp16(&Bt[buf][r * 64 + c8], W + (size_t)(nBase + r) * K + k0 + c8);
        }
    };

    v8f acc[2][4] = {};
    const int nIter = K >> 6;
    issueTiles(0, 0);
    for (int it = 0; it < nIter; ++it) {
        int cur = it & 1;
        if (it + 1 < nIter) { issueTiles((it + 1) << 6, cur ^ 1); wait_asynccnt<12>(); }
        else                { wait_asynccnt<0>(); }
        __syncthreads();
        #pragma unroll
        for (int ks = 0; ks < 64; ks += 32) {
            v16bf af0 = load_a_frag(At[cur], w * 32,      64, ks, lane);
            v16bf af1 = load_a_frag(At[cur], w * 32 + 16, 64, ks, lane);
            #pragma unroll
            for (int nt = 0; nt < 4; ++nt) {
                v16bf bfrag = load_b_frag(Bt[cur], nt * 16, 64, ks, lane);
                acc[0][nt] = wmma_bf16(af0, bfrag, acc[0][nt]);
                acc[1][nt] = wmma_bf16(af1, bfrag, acc[1][nt]);
            }
        }
        __syncthreads();
    }

    #pragma unroll
    for (int half = 0; half < 2; ++half) {
        #pragma unroll
        for (int nt = 0; nt < 4; ++nt) {
            #pragma unroll
            for (int r = 0; r < 8; ++r) {
                int row = mBase + w * 32 + half * 16 + ((lane < 16) ? r : r + 8);
                int col = nBase + nt * 16 + (lane & 15);
                float v = acc[half][nt][r];
                if (BIAS) v += bias[col];
                if (GELU) v  = gelu_tanh(v);
                if (RES)  v += residual[(size_t)row * N + col];
                if (OUTBF) Cb[(size_t)row * N + col] = f2bf(v);
                else       C [(size_t)row * N + col] = v;
            }
        }
    }
}

// --------------------------------------------- block-causal flash attention -
// grid = (SEQ/BLOCK, HEADS, BATCH), 128 threads (4 waves x 16 query rows).
// bf16 Q/K/V in [b][s][h*HDIM+d]; scale applied to fp32 S accumulator.

__global__ __launch_bounds__(128) void attn_kernel(
    const __bf16* __restrict__ Q, const __bf16* __restrict__ Kx,
    const __bf16* __restrict__ V, __bf16* __restrict__ O)
{
    __shared__ alignas(64) __bf16 Qt [64 * 64];
    __shared__ alignas(64) __bf16 Kt [64 * 64];
    __shared__ alignas(64) __bf16 VTt[64 * 64];   // V transposed: [d][n]
    __shared__ alignas(64) __bf16 Pt [64 * 64];
    int tid = threadIdx.x, lane = tid & 31, w = tid >> 5;
    int qb = blockIdx.x, h = blockIdx.y, b = blockIdx.z;
    const float scale = 0.125f;                    // 1/sqrt(HDIM)
    const int LD = HEADS * HDIM;                   // 512
    size_t baseQ = ((size_t)b * SEQ + (size_t)qb * 64) * LD + h * HDIM;

    #pragma unroll
    for (int v = 0; v < 4; ++v) {                  // async Q tile (raw bf16)
        int e = tid + v * 128; int r = e >> 3; int c8 = (e & 7) << 3;
        cp16(&Qt[r * 64 + c8], Q + baseQ + (size_t)r * LD + c8);
    }

    float mi[8], li[8];
    #pragma unroll
    for (int r = 0; r < 8; ++r) { mi[r] = -1e30f; li[r] = 0.f; }
    v8f accO[4] = {};

    for (int kb = 0; kb <= qb; ++kb) {
        size_t baseK = ((size_t)b * SEQ + (size_t)kb * 64) * LD + h * HDIM;
        #pragma unroll
        for (int v = 0; v < 4; ++v) {              // async K tile
            int e = tid + v * 128; int r = e >> 3; int c8 = (e & 7) << 3;
            cp16(&Kt[r * 64 + c8], Kx + baseK + (size_t)r * LD + c8);
        }
        #pragma unroll
        for (int v = 0; v < 4; ++v) {              // V tile, transposed in LDS
            int e = tid + v * 128; int r = e >> 3; int c8 = (e & 7) << 3;
            union { uint4 u; __bf16 b[8]; } tv;
            tv.u = *(const uint4*)(V + baseK + (size_t)r * LD + c8);
            #pragma unroll
            for (int j = 0; j < 8; ++j) VTt[(c8 + j) * 64 + r] = tv.b[j];
        }
        wait_asynccnt<0>();
        __syncthreads();

        // S = Q @ K^T
        v8f accS[4] = {};
        #pragma unroll
        for (int ks = 0; ks < 64; ks += 32) {
            v16bf af = load_a_frag(Qt, w * 16, 64, ks, lane);
            #pragma unroll
            for (int nt = 0; nt < 4; ++nt) {
                v16bf bfrag = load_b_frag(Kt, nt * 16, 64, ks, lane);
                accS[nt] = wmma_bf16(af, bfrag, accS[nt]);
            }
        }

        // online softmax (rows: lane<16 -> r, lane>=16 -> r+8)
        #pragma unroll
        for (int r = 0; r < 8; ++r) {
            float s0 = accS[0][r] * scale, s1 = accS[1][r] * scale;
            float s2 = accS[2][r] * scale, s3 = accS[3][r] * scale;
            float rmax = fmaxf(fmaxf(s0, s1), fmaxf(s2, s3));
            #pragma unroll
            for (int off = 8; off >= 1; off >>= 1)
                rmax = fmaxf(rmax, __shfl_xor(rmax, off, 32));
            float mnew  = fmaxf(mi[r], rmax);
            float alpha = __expf(mi[r] - mnew);
            mi[r] = mnew;
            li[r] *= alpha;
            #pragma unroll
            for (int dt = 0; dt < 4; ++dt) accO[dt][r] *= alpha;
            int row = w * 16 + ((lane < 16) ? r : r + 8);
            float p0 = __expf(s0 - mnew), p1 = __expf(s1 - mnew);
            float p2 = __expf(s2 - mnew), p3 = __expf(s3 - mnew);
            int cl = lane & 15;
            Pt[row * 64 +  0 + cl] = f2bf(p0);
            Pt[row * 64 + 16 + cl] = f2bf(p1);
            Pt[row * 64 + 32 + cl] = f2bf(p2);
            Pt[row * 64 + 48 + cl] = f2bf(p3);
            float rs = p0 + p1 + p2 + p3;
            #pragma unroll
            for (int off = 8; off >= 1; off >>= 1)
                rs += __shfl_xor(rs, off, 32);
            li[r] += rs;
        }
        __syncthreads();

        // O += P @ V   (B = V^T in LDS)
        #pragma unroll
        for (int ks = 0; ks < 64; ks += 32) {
            v16bf af = load_a_frag(Pt, w * 16, 64, ks, lane);
            #pragma unroll
            for (int dt = 0; dt < 4; ++dt) {
                v16bf bfrag = load_b_frag(VTt, dt * 16, 64, ks, lane);
                accO[dt] = wmma_bf16(af, bfrag, accO[dt]);
            }
        }
        __syncthreads();
    }

    #pragma unroll
    for (int dt = 0; dt < 4; ++dt) {
        #pragma unroll
        for (int r = 0; r < 8; ++r) {
            int row = w * 16 + ((lane < 16) ? r : r + 8);
            int col = dt * 16 + (lane & 15);
            O[baseQ + (size_t)row * LD + col] = f2bf(accO[dt][r] / li[r]);
        }
    }
}

// ----------------------------------------------------------------- launch ---

static inline void conv_bf16(const float* src, void* dst, size_t n, hipStream_t s) {
    int n4 = (int)(n / 4);
    int blocks = (n4 + 255) / 256; if (blocks > 4096) blocks = 4096;
    convert_bf16_kernel<<<blocks, 256, 0, s>>>(src, (unsigned int*)dst, n4);
}

extern "C" void kernel_launch(void* const* d_in, const int* in_sizes, int n_in,
                              void* d_out, int out_size, void* d_ws, size_t ws_size,
                              hipStream_t stream) {
    const int*   seq = (const int*)  d_in[0];
    const float* emb = (const float*)d_in[1];
    const float* wq  = (const float*)d_in[2];
    const float* wk  = (const float*)d_in[3];
    const float* wv  = (const float*)d_in[4];
    const float* wo  = (const float*)d_in[5];
    const float* ffw = (const float*)d_in[6];
    const float* ffb = (const float*)d_in[7];
    const float* n1w = (const float*)d_in[8];
    const float* n1b = (const float*)d_in[9];
    const float* n2w = (const float*)d_in[10];
    const float* n2b = (const float*)d_in[11];
    const float* fnw = (const float*)d_in[12];
    const float* fnb = (const float*)d_in[13];
    const float* hw  = (const float*)d_in[14];
    const float* hb  = (const float*)d_in[15];
    float* out = (float*)d_out;

    const size_t WEL = (size_t)DIM * DIM;          // 262144
    char* base = (char*)d_ws;
    float*  x   = (float*) (base);                 // 8 MiB fp32 residual stream
    __bf16* hbf = (__bf16*)(base + (8ull  << 20)); // 8..12 MiB
    __bf16* qbf = (__bf16*)(base + (12ull << 20));
    __bf16* kbf = (__bf16*)(base + (16ull << 20));
    __bf16* vbf = (__bf16*)(base + (20ull << 20));
    __bf16* obf = (__bf16*)(base + (24ull << 20));
    __bf16* wsc = (__bf16*)(base + (28ull << 20)); // 32 MiB weight scratch

    dim3 gemmGrid(DIM / 64, MROWS / 128);          // (8, 32)
    dim3 headGrid(VOCAB / 64, MROWS / 128);        // (500, 32)
    dim3 attnGrid(SEQ / BLOCK, HEADS, BATCH);      // (32, 8, 2)

    embed_pe_kernel<<<MROWS, 256, 0, stream>>>(seq, emb, x);

    for (int l = 0; l < DEPTH; ++l) {
        __bf16 *Wqb = wsc, *Wkb = wsc + WEL, *Wvb = wsc + 2 * WEL;
        __bf16 *Wob = wsc + 3 * WEL, *Wfb = wsc + 4 * WEL;
        conv_bf16(wq  + l * WEL, Wqb, WEL, stream);
        conv_bf16(wk  + l * WEL, Wkb, WEL, stream);
        conv_bf16(wv  + l * WEL, Wvb, WEL, stream);
        conv_bf16(wo  + l * WEL, Wob, WEL, stream);
        conv_bf16(ffw + l * WEL, Wfb, WEL, stream);

        rmsnorm_kernel<<<MROWS, 256, 0, stream>>>(x, n1w + l * DIM, n1b + l * DIM, hbf);
        gemm_bf16_kernel<false,false,false,true><<<gemmGrid, 128, 0, stream>>>(
            hbf, Wqb, nullptr, nullptr, nullptr, qbf, DIM, DIM);
        gemm_bf16_kernel<false,false,false,true><<<gemmGrid, 128, 0, stream>>>(
            hbf, Wkb, nullptr, nullptr, nullptr, kbf, DIM, DIM);
        gemm_bf16_kernel<false,false,false,true><<<gemmGrid, 128, 0, stream>>>(
            hbf, Wvb, nullptr, nullptr, nullptr, vbf, DIM, DIM);
        attn_kernel<<<attnGrid, 128, 0, stream>>>(qbf, kbf, vbf, obf);
        gemm_bf16_kernel<false,false,true,false><<<gemmGrid, 128, 0, stream>>>(
            obf, Wob, nullptr, x, x, nullptr, DIM, DIM);
        rmsnorm_kernel<<<MROWS, 256, 0, stream>>>(x, n2w + l * DIM, n2b + l * DIM, hbf);
        gemm_bf16_kernel<true,true,true,false><<<gemmGrid, 128, 0, stream>>>(
            hbf, Wfb, ffb + l * DIM, x, x, nullptr, DIM, DIM);
    }

    rmsnorm_kernel<<<MROWS, 256, 0, stream>>>(x, fnw, fnb, hbf);
    conv_bf16(hw, wsc, (size_t)VOCAB * DIM, stream);
    gemm_bf16_kernel<true,false,false,false><<<headGrid, 128, 0, stream>>>(
        hbf, wsc, hb, nullptr, out, nullptr, VOCAB, DIM);
}
